// Qwen3NextExpertsForEngine_32392643347144
// MI455X (gfx1250) — matrix operand order repaired and checked by reading, not compiled, not run on hardware
//
#include <hip/hip_runtime.h>

// ---------------------------------------------------------------------------
// Qwen3-Next MoE experts for MI455X (gfx1250, wave32, WMMA bf16 16x16x32)
// Double-buffered LDS pipeline; async global->LDS copy for the bf16 h operand;
// f32->bf16 via v_cvt_pk_bf16_f32; sched_barrier pins load-issue before WMMA.
// ---------------------------------------------------------------------------
typedef __attribute__((ext_vector_type(16))) __bf16 v16bf;
typedef __attribute__((ext_vector_type(8)))  float  v8f;

#define T_TOK   4096
#define TOPK    8
#define NEXP    64
#define HID     2048
#define INTER   512
#define NPAIR   (T_TOK * TOPK)            // 32768
#define TILE_M  64
#define MAX_TILES (NPAIR / TILE_M + NEXP) // 576 upper bound on sum ceil(cnt/64)
#define LDSS    40                        // ushort stride per LDS row (32 data + 8 pad)

union FragBF { uint4 u[2]; v16bf v; };

__device__ __forceinline__ void sched_fence() {
#if __has_builtin(__builtin_amdgcn_sched_barrier)
    __builtin_amdgcn_sched_barrier(0);    // nothing crosses: pin issue order
#endif
}

__device__ __forceinline__ unsigned short f2bf(float f) {
    __bf16 h = (__bf16)f;
    return __builtin_bit_cast(unsigned short, h);
}
// one VALU op converting+packing 2 floats -> 2 bf16 (dst.lo = a, dst.hi = b)
__device__ __forceinline__ unsigned int pack2(float a, float b) {
    unsigned int r;
    asm("v_cvt_pk_bf16_f32 %0, %1, %2" : "=v"(r) : "v"(a), "v"(b));
    return r;
}
__device__ __forceinline__ uint4 cvt8(float4 a, float4 b) {
    uint4 o;
    o.x = pack2(a.x, a.y); o.y = pack2(a.z, a.w);
    o.z = pack2(b.x, b.y); o.w = pack2(b.z, b.w);
    return o;
}

// generic pointer low 32 bits == LDS byte address
#define LDS_OFF(p) ((unsigned)(size_t)(p))

// async 16-byte global->LDS copy per lane (ASYNCcnt-tracked)
__device__ __forceinline__ void async_ld16(unsigned ldsAddr, const void* gaddr) {
    asm volatile("global_load_async_to_lds_b128 %0, %1, off"
                 :: "v"(ldsAddr), "v"(gaddr) : "memory");
}
__device__ __forceinline__ void wait_async0() {
    asm volatile("s_wait_asynccnt 0x0" ::: "memory");
}

// ------------------------------- routing -----------------------------------
__global__ void k_init(int* counts, int* cursor, int* numTiles) {
    int t = threadIdx.x;
    if (t < NEXP) { counts[t] = 0; cursor[t] = 0; }
    if (t == 0) numTiles[0] = 0;
}

__global__ void k_zero_out(float4* out) {
    out[(size_t)blockIdx.x * blockDim.x + threadIdx.x] = make_float4(0.f, 0.f, 0.f, 0.f);
}

__global__ void k_count(const int* __restrict__ tki, int* counts) {
    int p = blockIdx.x * blockDim.x + threadIdx.x;
    if (p < NPAIR) atomicAdd(&counts[tki[p]], 1);
}

__global__ void k_build(const int* __restrict__ counts, int* offsets,
                        int* tileE, int* tileR, int* numTiles) {
    if (threadIdx.x == 0 && blockIdx.x == 0) {
        int off = 0;
        for (int e = 0; e < NEXP; ++e) { offsets[e] = off; off += counts[e]; }
        offsets[NEXP] = off;
        int t = 0;
        for (int e = 0; e < NEXP; ++e)
            for (int r = 0; r < counts[e]; r += TILE_M) {
                tileE[t] = e; tileR[t] = offsets[e] + r; ++t;
            }
        numTiles[0] = t;
    }
}

__global__ void k_scatter(const int* __restrict__ tki, const int* __restrict__ offsets,
                          int* cursor, int* sortedPair) {
    int p = blockIdx.x * blockDim.x + threadIdx.x;
    if (p < NPAIR) {
        int e = tki[p];
        int pos = offsets[e] + atomicAdd(&cursor[e], 1);
        sortedPair[pos] = p;
    }
}

// ------------------------------- GEMM 1 ------------------------------------
// gate_up = x @ Wgu^T per expert, h = silu(gate)*up -> hws (bf16)
#define G1_COMPUTE(BUF)                                                              \
    {                                                                                \
        FragBF a_, b_;                                                               \
        const unsigned short* ar_ = &lA[BUF][aOff];                                  \
        a_.u[0] = *(const uint4*)&ar_[half * 16];                                    \
        a_.u[1] = *(const uint4*)&ar_[half * 16 + 8];                                \
        const unsigned short* p_;                                                    \
        p_ = &lBg[BUF][bOff0]; b_.u[0] = *(const uint4*)&p_[0]; b_.u[1] = *(const uint4*)&p_[8]; \
        ag0 = __builtin_amdgcn_wmma_f32_16x16x32_bf16(false, a_.v, false, b_.v, (short)0, ag0, false, false); \
        p_ = &lBg[BUF][bOff1]; b_.u[0] = *(const uint4*)&p_[0]; b_.u[1] = *(const uint4*)&p_[8]; \
        ag1 = __builtin_amdgcn_wmma_f32_16x16x32_bf16(false, a_.v, false, b_.v, (short)0, ag1, false, false); \
        p_ = &lBu[BUF][bOff0]; b_.u[0] = *(const uint4*)&p_[0]; b_.u[1] = *(const uint4*)&p_[8]; \
        au0 = __builtin_amdgcn_wmma_f32_16x16x32_bf16(false, a_.v, false, b_.v, (short)0, au0, false, false); \
        p_ = &lBu[BUF][bOff1]; b_.u[0] = *(const uint4*)&p_[0]; b_.u[1] = *(const uint4*)&p_[8]; \
        au1 = __builtin_amdgcn_wmma_f32_16x16x32_bf16(false, a_.v, false, b_.v, (short)0, au1, false, false); \
    }

__global__ __launch_bounds__(256)
void k_gemm1(const float* __restrict__ hidden, const float* __restrict__ gup,
             const int* __restrict__ sortedPair, const int* __restrict__ offsets,
             const int* __restrict__ tileE, const int* __restrict__ tileR,
             const int* __restrict__ numTiles, unsigned short* __restrict__ hws) {
    __shared__ unsigned short lA [2][TILE_M * LDSS];
    __shared__ unsigned short lBg[2][TILE_M * LDSS];
    __shared__ unsigned short lBu[2][TILE_M * LDSS];

    const int tile = blockIdx.x;
    if (tile >= numTiles[0]) return;
    const int e      = tileE[tile];
    const int row0   = tileR[tile];
    const int rowEnd = offsets[e + 1];
    const int n0     = blockIdx.y * 64;          // h columns [n0, n0+64)

    const int tid  = threadIdx.x;
    const int lrow = tid >> 2;                   // 0..63
    const int kg   = (tid & 3) * 8;              // 0,8,16,24
    const int sIdx = lrow * LDSS + kg;

    const int srow = row0 + lrow;
    const int gr   = (srow < rowEnd) ? srow : row0;
    const int pr   = sortedPair[gr];
    const float* aRow  = hidden + (size_t)(pr >> 3) * HID;
    const float* bgRow = gup + ((size_t)e * (2 * INTER) + (n0 + lrow)) * HID;
    const float* buRow = bgRow + (size_t)INTER * HID;

    const int lane = tid & 31;
    const int wid  = tid >> 5;
    const int wm   = (wid & 3) * 16;             // wave M offset in tile
    const int wn   = (wid >> 2) * 32;            // wave N offset in tile
    const int half = lane >> 4;
    const int lsub = lane & 15;
    const int aOff  = (wm + lsub) * LDSS;
    const int bOff0 = (wn + lsub) * LDSS + half * 16;
    const int bOff1 = (wn + 16 + lsub) * LDSS + half * 16;

    v8f ag0 = {}, ag1 = {}, au0 = {}, au1 = {};

    // prologue: stage kb=0 into buffer 0
    {
        const float4* pa = (const float4*)(aRow + kg);
        const float4* pg = (const float4*)(bgRow + kg);
        const float4* pu = (const float4*)(buRow + kg);
        *(uint4*)&lA [0][sIdx] = cvt8(pa[0], pa[1]);
        *(uint4*)&lBg[0][sIdx] = cvt8(pg[0], pg[1]);
        *(uint4*)&lBu[0][sIdx] = cvt8(pu[0], pu[1]);
    }
    __syncthreads();

    int cur = 0;
    for (int kb = 32; kb < HID; kb += 32) {
        // issue next K-slice global loads first, pinned before the WMMA block
        const float4* pa = (const float4*)(aRow + kb + kg);
        const float4* pg = (const float4*)(bgRow + kb + kg);
        const float4* pu = (const float4*)(buRow + kb + kg);
        float4 a0 = pa[0], a1 = pa[1];
        float4 g0 = pg[0], g1 = pg[1];
        float4 u0 = pu[0], u1 = pu[1];
        sched_fence();   // loads issue here; WMMAs below cover their latency

        G1_COMPUTE(cur);

        const int nxt = cur ^ 1;
        *(uint4*)&lA [nxt][sIdx] = cvt8(a0, a1);
        *(uint4*)&lBg[nxt][sIdx] = cvt8(g0, g1);
        *(uint4*)&lBu[nxt][sIdx] = cvt8(u0, u1);
        __syncthreads();
        cur = nxt;
    }
    G1_COMPUTE(cur);

    // epilogue: h = silu(gate)*up, stored bf16. C/D layout: VGPR i -> M=i+8*half
    for (int ns = 0; ns < 2; ++ns) {
        v8f g = ns ? ag1 : ag0;
        v8f u = ns ? au1 : au0;
        const int col = n0 + wn + ns * 16 + lsub;
        for (int i = 0; i < 8; ++i) {
            const int r = row0 + wm + i + half * 8;
            if (r < rowEnd) {
                float gv = g[i];
                float hv = (gv / (1.0f + __expf(-gv))) * u[i];
                hws[(size_t)r * INTER + col] = f2bf(hv);
            }
        }
    }
}

// ------------------------------- GEMM 2 ------------------------------------
// y = h @ Wd^T, scale by routing weight, scatter-add into out[token]
#define G2_COMPUTE(BUF)                                                              \
    {                                                                                \
        FragBF a_, b_;                                                               \
        const unsigned short* ar_ = &lA[BUF][aOff];                                  \
        a_.u[0] = *(const uint4*)&ar_[half * 16];                                    \
        a_.u[1] = *(const uint4*)&ar_[half * 16 + 8];                                \
        const unsigned short* p_;                                                    \
        p_ = &lB[BUF][bOff0]; b_.u[0] = *(const uint4*)&p_[0]; b_.u[1] = *(const uint4*)&p_[8]; \
        ac0 = __builtin_amdgcn_wmma_f32_16x16x32_bf16(false, a_.v, false, b_.v, (short)0, ac0, false, false); \
        p_ = &lB[BUF][bOff1]; b_.u[0] = *(const uint4*)&p_[0]; b_.u[1] = *(const uint4*)&p_[8]; \
        ac1 = __builtin_amdgcn_wmma_f32_16x16x32_bf16(false, a_.v, false, b_.v, (short)0, ac1, false, false); \
    }

__global__ __launch_bounds__(256)
void k_gemm2(const unsigned short* __restrict__ hws, const float* __restrict__ down,
             const float* __restrict__ tkw,
             const int* __restrict__ sortedPair, const int* __restrict__ offsets,
             const int* __restrict__ tileE, const int* __restrict__ tileR,
             const int* __restrict__ numTiles, float* __restrict__ out) {
    __shared__ unsigned short lA[2][TILE_M * LDSS];
    __shared__ unsigned short lB[2][TILE_M * LDSS];

    const int tile = blockIdx.x;
    if (tile >= numTiles[0]) return;
    const int e      = tileE[tile];
    const int row0   = tileR[tile];
    const int rowEnd = offsets[e + 1];
    const int n0     = blockIdx.y * 64;          // out columns [n0, n0+64)

    const int tid  = threadIdx.x;
    const int lrow = tid >> 2;
    const int kg   = (tid & 3) * 8;
    const int sIdx = lrow * LDSS + kg;

    const int gr = min(row0 + lrow, rowEnd - 1);
    const unsigned short* aRow = hws + (size_t)gr * INTER;   // bf16 rows of h
    const float* bRow = down + ((size_t)e * HID + (n0 + lrow)) * INTER;

    const int lane = tid & 31;
    const int wid  = tid >> 5;
    const int wm   = (wid & 3) * 16;
    const int wn   = (wid >> 2) * 32;
    const int half = lane >> 4;
    const int lsub = lane & 15;
    const int aOff  = (wm + lsub) * LDSS;
    const int bOff0 = (wn + lsub) * LDSS + half * 16;
    const int bOff1 = (wn + 16 + lsub) * LDSS + half * 16;

    v8f ac0 = {}, ac1 = {};

    // prologue: A via async global->LDS (already bf16), B via convert
    async_ld16(LDS_OFF(&lA[0][sIdx]), aRow + kg);
    {
        const float4* pb = (const float4*)(bRow + kg);
        *(uint4*)&lB[0][sIdx] = cvt8(pb[0], pb[1]);
    }
    wait_async0();
    __syncthreads();

    int cur = 0;
    for (int kb = 32; kb < INTER; kb += 32) {
        const int nxt = cur ^ 1;
        // next A slice: raw bf16 async copy into the other buffer
        async_ld16(LDS_OFF(&lA[nxt][sIdx]), aRow + kb + kg);
        // next B slice to registers
        const float4* pb = (const float4*)(bRow + kb + kg);
        float4 b0 = pb[0], b1 = pb[1];
        sched_fence();   // async + loads issue before the WMMA pair

        G2_COMPUTE(cur);

        *(uint4*)&lB[nxt][sIdx] = cvt8(b0, b1);
        wait_async0();
        __syncthreads();
        cur = nxt;
    }
    G2_COMPUTE(cur);

    for (int ns = 0; ns < 2; ++ns) {
        v8f acc = ns ? ac1 : ac0;
        const int col = n0 + wn + ns * 16 + lsub;
        for (int i = 0; i < 8; ++i) {
            const int r = row0 + wm + i + half * 8;
            if (r < rowEnd) {
                const int pr = sortedPair[r];
                const float w = tkw[pr];
                atomicAdd(&out[(size_t)(pr >> 3) * HID + col], acc[i] * w);
            }
        }
    }
}

// ------------------------------- launch ------------------------------------
extern "C" void kernel_launch(void* const* d_in, const int* in_sizes, int n_in,
                              void* d_out, int out_size, void* d_ws, size_t ws_size,
                              hipStream_t stream) {
    const float* hidden = (const float*)d_in[0];  // [T, H]
    const int*   tki    = (const int*)  d_in[1];  // [T, K]
    const float* tkw    = (const float*)d_in[2];  // [T, K]
    const float* gup    = (const float*)d_in[3];  // [E, 2I, H]
    const float* down   = (const float*)d_in[4];  // [E, H, I]
    float* out = (float*)d_out;                   // [T, H]

    int* wsI       = (int*)d_ws;
    int* counts    = wsI;            // 64
    int* cursor    = wsI + 64;       // 64
    int* offsets   = wsI + 128;      // 65
    int* numTiles  = wsI + 256;      // 1
    int* tileE     = wsI + 512;      // MAX_TILES
    int* tileR     = wsI + 1536;     // MAX_TILES
    int* sortedPr  = wsI + 2560;     // NPAIR
    unsigned short* hws = (unsigned short*)((char*)d_ws + (size_t)(2560 + NPAIR) * sizeof(int));

    k_init    <<<1, 64, 0, stream>>>(counts, cursor, numTiles);
    k_zero_out<<<(T_TOK * HID / 4) / 256, 256, 0, stream>>>((float4*)out);
    k_count   <<<NPAIR / 256, 256, 0, stream>>>(tki, counts);
    k_build   <<<1, 1, 0, stream>>>(counts, offsets, tileE, tileR, numTiles);
    k_scatter <<<NPAIR / 256, 256, 0, stream>>>(tki, offsets, cursor, sortedPr);

    dim3 g1(MAX_TILES, INTER / 64);  // 576 x 8
    k_gemm1<<<g1, 256, 0, stream>>>(hidden, gup, sortedPr, offsets, tileE, tileR, numTiles, hws);
    dim3 g2(MAX_TILES, HID / 64);    // 576 x 32
    k_gemm2<<<g2, 256, 0, stream>>>(hws, down, tkw, sortedPr, offsets, tileE, tileR, numTiles, out);
}
